// SlidingWindowAttention_25099788877909
// MI455X (gfx1250) — matrix operand order repaired and checked by reading, not compiled, not run on hardware
//
#include <hip/hip_runtime.h>

#define BATCH 2
#define SEQ 2048
#define DM 2048
#define NH 16
#define HD 128

typedef __attribute__((ext_vector_type(16))) __bf16 v16bf;
typedef __attribute__((ext_vector_type(8)))  __bf16 v8bf;
typedef __attribute__((ext_vector_type(8)))  float  v8f;
typedef unsigned int u32x4 __attribute__((ext_vector_type(4)));
typedef int          i32x8 __attribute__((ext_vector_type(8)));
typedef int          i32x4 __attribute__((ext_vector_type(4)));

#if __has_builtin(__builtin_amdgcn_tensor_load_to_lds)
#define HAVE_TDM 1
#else
#define HAVE_TDM 0
#endif

__device__ __forceinline__ v16bf join8(v8bf a, v8bf b) {
  return __builtin_shufflevector(a, b, 0,1,2,3,4,5,6,7,8,9,10,11,12,13,14,15);
}

// ---------------------------------------------------------------------------
// fp32 -> bf16 elementwise convert
// ---------------------------------------------------------------------------
__global__ __launch_bounds__(256)
void f32_to_bf16(const float* __restrict__ in, __bf16* __restrict__ out, int n) {
  int i = blockIdx.x * blockDim.x + threadIdx.x;
  if (i < n) out[i] = (__bf16)in[i];
}

// ---------------------------------------------------------------------------
// RoPE: read fp32 [B*S, DM], rotate pairs within each head, write bf16
// head-major [B, H, S, HD]. One thread per (even,odd) pair.
// ---------------------------------------------------------------------------
__global__ __launch_bounds__(256)
void rope_to_bf16(const float* __restrict__ in, __bf16* __restrict__ out) {
  int i = blockIdx.x * blockDim.x + threadIdx.x;   // pair index, < B*S*DM/2
  int pr = i & 63;              // pair index within head (0..63)
  int h  = (i >> 6) & (NH - 1);
  int s  = (i >> 10) & (SEQ - 1);
  int b  = i >> 21;
  float inv_freq = __powf(10000.0f, -(2.0f * (float)pr) / (float)HD);
  float ang = (float)s * inv_freq;
  float c = __cosf(ang), sn = __sinf(ang);
  const float* src = in + ((size_t)(b * SEQ + s)) * DM + h * HD + 2 * pr;
  float x1 = src[0], x2 = src[1];
  __bf16* dst = out + (((size_t)(b * NH + h) * SEQ + s)) * HD + 2 * pr;
  dst[0] = (__bf16)(x1 * c - x2 * sn);
  dst[1] = (__bf16)(x1 * sn + x2 * c);
}

// ---------------------------------------------------------------------------
// V: fp32 [B*S, DM] -> bf16 transposed [B, H, HD, S]
// ---------------------------------------------------------------------------
__global__ __launch_bounds__(256)
void v_transpose_bf16(const float* __restrict__ in, __bf16* __restrict__ out) {
  int i = blockIdx.x * blockDim.x + threadIdx.x;   // < B*S*DM
  int d = i & (HD - 1);
  int h = (i >> 7) & (NH - 1);
  int s = (i >> 11) & (SEQ - 1);
  int b = i >> 22;
  out[(((size_t)(b * NH + h) * HD + d)) * SEQ + s] = (__bf16)in[i];
}

// ---------------------------------------------------------------------------
// TDM: DMA a 64-row x 128-element bf16 slab of B (row-major [N,K]) into LDS.
// D# per CDNA5 ISA ch.8: group0 = {count/type/addr}, group1 = {dims/strides}.
// Issued by one wave; TDM ignores EXEC. Completion via TENSORcnt.
// ---------------------------------------------------------------------------
#if HAVE_TDM
__device__ __forceinline__ void tdm_load_b_tile(const __bf16* B, int n0, int kc,
                                                int K, int N,
                                                const __bf16* ldsdst) {
  unsigned long long ga =
      (unsigned long long)(uintptr_t)(B + (size_t)n0 * K + kc);
  unsigned int loff = (unsigned int)(uintptr_t)ldsdst;  // LDS byte offset

  u32x4 g0;
  g0.x = 1u;                                  // count=1 (valid user D#)
  g0.y = loff;                                // lds_addr
  g0.z = (unsigned int)(ga & 0xFFFFFFFFu);    // global_addr[31:0]
  g0.w = (unsigned int)((ga >> 32) & 0x01FFFFFFu) | (2u << 30);  // [56:32]|type=2

  i32x8 g1;
  g1[0] = (int)(1u << 16);                    // wg_mask=0, data_size=1 (2 bytes)
  g1[1] = (int)(((unsigned)K & 0xFFFFu) << 16);               // tensor_dim0 lo16
  g1[2] = (int)((((unsigned)K >> 16) & 0xFFFFu)
                | (((unsigned)N & 0xFFFFu) << 16));           // dim0 hi | dim1 lo
  g1[3] = (int)((((unsigned)N >> 16) & 0xFFFFu) | (128u << 16)); // dim1 hi | tile_dim0=128
  g1[4] = 64;                                 // tile_dim1=64, tile_dim2=0
  g1[5] = K;                                  // tensor_dim0_stride lo32 (elements)
  g1[6] = 0;                                  // stride0 hi16 | stride1 lo16
  g1[7] = 0;
  i32x4 gz4 = {0, 0, 0, 0};                   // 2-D tensor: groups 2/3 unused
  i32x8 gz8 = {0, 0, 0, 0, 0, 0, 0, 0};       // extra group (clang-23 6-arg form)
  __builtin_amdgcn_tensor_load_to_lds(g0, g1, gz4, gz4, gz8, 0);
}
#endif

// ---------------------------------------------------------------------------
// bf16 NT GEMM: C[M,N] = A[M,K] * B[N,K]^T, fp32 accumulate/output.
// Block = 128 threads (4 waves). Workgroup tile 128x64. B slabs (64x128)
// staged in LDS: TDM double-buffered (TENSORcnt) when available, else a
// cooperative copy. 32 WMMAs per 128-k chunk per wave.
// ---------------------------------------------------------------------------
__global__ __launch_bounds__(128)
void gemm_bf16_nt(const __bf16* __restrict__ A, const __bf16* __restrict__ B,
                  float* __restrict__ C, int M, int N, int K) {
  __shared__ __bf16 Bs[2][64 * 128];          // 2 x 16 KB

  const int lane = threadIdx.x & 31;
  const int wave = threadIdx.x >> 5;
  const int l16  = lane & 15;
  const int half = lane >> 4;
  const int m0 = blockIdx.y * 128 + wave * 32;
  const int n0 = blockIdx.x * 64;
  const int kA = half * 16;   // A-fragment per-lane K sub-offset
  const int kB = half * 16;   // B-fragment per-lane K sub-offset

  v8f acc[2][4] = {};

  const __bf16* arow0 = A + (size_t)(m0 + l16) * K;
  const __bf16* arow1 = A + (size_t)(m0 + 16 + l16) * K;

#if HAVE_TDM
  if (wave == 0) tdm_load_b_tile(B, n0, 0, K, N, &Bs[0][0]);
#endif

  int buf = 0;
  for (int kc = 0; kc < K; kc += 128, buf ^= 1) {
#if HAVE_TDM
    if (wave == 0) {
      if (kc + 128 < K) {
        tdm_load_b_tile(B, n0, kc + 128, K, N, &Bs[buf ^ 1][0]);
        __builtin_amdgcn_s_wait_tensorcnt(1);   // oldest (current tile) done
      } else {
        __builtin_amdgcn_s_wait_tensorcnt(0);
      }
    }
    __syncthreads();
    const __bf16* Bt = &Bs[buf][0];
#else
    __syncthreads();
    // cooperative copy: 128 threads x 64 bf16 each
    {
      int row = threadIdx.x >> 1;
      int hcol = (threadIdx.x & 1) * 64;
      const __bf16* src = B + (size_t)(n0 + row) * K + kc + hcol;
      __bf16* dst = &Bs[0][row * 128 + hcol];
#pragma unroll
      for (int j = 0; j < 8; ++j)
        *(v8bf*)(dst + j * 8) = *(const v8bf*)(src + j * 8);
    }
    __syncthreads();
    const __bf16* Bt = &Bs[0][0];
#endif

#pragma unroll
    for (int kk = 0; kk < 128; kk += 32) {
      v16bf a0 = join8(*(const v8bf*)(arow0 + kc + kk + kA),
                       *(const v8bf*)(arow0 + kc + kk + 8 + kA));
      v16bf a1 = join8(*(const v8bf*)(arow1 + kc + kk + kA),
                       *(const v8bf*)(arow1 + kc + kk + 8 + kA));
#pragma unroll
      for (int t = 0; t < 4; ++t) {
        v16bf bf = *(const v16bf*)(Bt + (size_t)(t * 16 + l16) * 128 + kk + kB);
        acc[0][t] = __builtin_amdgcn_wmma_f32_16x16x32_bf16(
            false, a0, false, bf, (short)0, acc[0][t], false, false);
        acc[1][t] = __builtin_amdgcn_wmma_f32_16x16x32_bf16(
            false, a1, false, bf, (short)0, acc[1][t], false, false);
      }
    }
    __syncthreads();
  }

#pragma unroll
  for (int i = 0; i < 2; ++i)
#pragma unroll
    for (int t = 0; t < 4; ++t)
#pragma unroll
      for (int r = 0; r < 8; ++r) {
        int row = m0 + i * 16 + r + 8 * half;
        int col = n0 + t * 16 + l16;
        C[(size_t)row * N + col] = acc[i][t][r];
      }
}

// ---------------------------------------------------------------------------
// Flash attention: one wave per (16 q-rows, batch-head). Online softmax,
// causal mask, Q fragments resident, P relayout C->A via 1KB LDS tile.
// Qb/Kb: [B,H,S,HD] bf16 (RoPE applied). Vt: [B,H,HD,S] bf16.
// Ob: [B*S, DM] bf16.
// ---------------------------------------------------------------------------
__global__ __launch_bounds__(32)
void flash_attn(const __bf16* __restrict__ Qb, const __bf16* __restrict__ Kb,
                const __bf16* __restrict__ Vt, __bf16* __restrict__ Ob) {
  __shared__ __bf16 Plds[16 * 32];

  const int lane = threadIdx.x & 31;
  const int l16  = lane & 15;
  const int half = lane >> 4;
  const int q0 = blockIdx.x * 16;
  const int bh = blockIdx.y;          // b*NH + h
  const int h  = bh & (NH - 1);
  const int b  = bh >> 4;
  const int kA = half * 16;
  const int kB = half * 16;

  const __bf16* Qbase = Qb + ((size_t)bh * SEQ + q0) * HD;
  const __bf16* Kbase = Kb + (size_t)bh * SEQ * HD;
  const __bf16* Vbase = Vt + (size_t)bh * HD * SEQ;

  // Resident Q A-fragments: 4 chunks of K=32 over head dim 128
  v16bf qf[4];
#pragma unroll
  for (int c = 0; c < 4; ++c) {
    const __bf16* p = Qbase + (size_t)l16 * HD + c * 32 + kA;
    qf[c] = join8(*(const v8bf*)p, *(const v8bf*)(p + 8));
  }

  v8f o[8] = {};
  float mrow[8], lrow[8], alpha[8], p0[8], p1[8];
#pragma unroll
  for (int r = 0; r < 8; ++r) { mrow[r] = -1e30f; lrow[r] = 0.0f; }

  const float sc = 0.0883883476483184f;   // 1/sqrt(128)
  const int qmax = q0 + 15;

  for (int kb = 0; kb <= qmax; kb += 32) {
    // ---- scores: two 16x16 tiles (keys kb..kb+15, kb+16..kb+31) ----
    v8f s0 = {}, s1 = {};
#pragma unroll
    for (int c = 0; c < 4; ++c) {
      v16bf k0 = *(const v16bf*)(Kbase + (size_t)(kb + l16) * HD + c * 32 + kB);
      v16bf k1 = *(const v16bf*)(Kbase + (size_t)(kb + 16 + l16) * HD + c * 32 + kB);
      s0 = __builtin_amdgcn_wmma_f32_16x16x32_bf16(false, qf[c], false, k0,
                                                   (short)0, s0, false, false);
      s1 = __builtin_amdgcn_wmma_f32_16x16x32_bf16(false, qf[c], false, k1,
                                                   (short)0, s1, false, false);
    }

    // ---- causal mask + online softmax (row = r + 8*half, col = l16) ----
#pragma unroll
    for (int r = 0; r < 8; ++r) {
      int qrow = q0 + r + 8 * half;
      float v0 = s0[r] * sc;
      float v1 = s1[r] * sc;
      if (kb + l16 > qrow)      v0 = -1e30f;
      if (kb + 16 + l16 > qrow) v1 = -1e30f;
      float rm = fmaxf(v0, v1);
      rm = fmaxf(rm, __shfl_xor(rm, 1, 32));
      rm = fmaxf(rm, __shfl_xor(rm, 2, 32));
      rm = fmaxf(rm, __shfl_xor(rm, 4, 32));
      rm = fmaxf(rm, __shfl_xor(rm, 8, 32));
      float mnew = fmaxf(mrow[r], rm);
      float a  = __expf(mrow[r] - mnew);
      float e0 = __expf(v0 - mnew);
      float e1 = __expf(v1 - mnew);
      float rs = e0 + e1;
      rs += __shfl_xor(rs, 1, 32);
      rs += __shfl_xor(rs, 2, 32);
      rs += __shfl_xor(rs, 4, 32);
      rs += __shfl_xor(rs, 8, 32);
      lrow[r] = lrow[r] * a + rs;
      mrow[r] = mnew;
      alpha[r] = a;
      p0[r] = e0; p1[r] = e1;
    }

    // ---- rescale accumulators ----
#pragma unroll
    for (int t = 0; t < 8; ++t)
#pragma unroll
      for (int r = 0; r < 8; ++r) o[t][r] *= alpha[r];

    // ---- P: C-layout -> A-layout via LDS (16 rows x 32 keys, bf16) ----
#pragma unroll
    for (int r = 0; r < 8; ++r) {
      int row = r + 8 * half;
      Plds[row * 32 + l16]      = (__bf16)p0[r];
      Plds[row * 32 + 16 + l16] = (__bf16)p1[r];
    }
    asm volatile("s_wait_dscnt 0" ::: "memory");
    const __bf16* pr = &Plds[l16 * 32 + kA];
    v16bf pf = join8(*(const v8bf*)pr, *(const v8bf*)(pr + 8));

    // ---- PV: 8 output chunks of 16 along head dim ----
#pragma unroll
    for (int t = 0; t < 8; ++t) {
      v16bf vf = *(const v16bf*)(Vbase + (size_t)(t * 16 + l16) * SEQ + kb + kB);
      o[t] = __builtin_amdgcn_wmma_f32_16x16x32_bf16(false, pf, false, vf,
                                                     (short)0, o[t], false, false);
    }
  }

  // ---- epilogue: normalize and store to [B*S, DM] bf16 ----
#pragma unroll
  for (int t = 0; t < 8; ++t)
#pragma unroll
    for (int r = 0; r < 8; ++r) {
      int qrow = q0 + r + 8 * half;
      float val = o[t][r] / lrow[r];
      Ob[((size_t)b * SEQ + qrow) * DM + h * HD + t * 16 + l16] = (__bf16)val;
    }
}

// ---------------------------------------------------------------------------
extern "C" void kernel_launch(void* const* d_in, const int* in_sizes, int n_in,
                              void* d_out, int out_size, void* d_ws, size_t ws_size,
                              hipStream_t stream) {
  const float* x  = (const float*)d_in[0];
  const float* Wq = (const float*)d_in[1];
  const float* Wk = (const float*)d_in[2];
  const float* Wv = (const float*)d_in[3];
  const float* Wo = (const float*)d_in[4];
  float* out = (float*)d_out;

  const int NE_X = BATCH * SEQ * DM;   // 8388608
  const int NE_W = DM * DM;            // 4194304
  const int M = BATCH * SEQ;           // 4096

  char* ws = (char*)d_ws;
  size_t off = 0;
  auto carve = [&](size_t bytes) { void* p = ws + off; off += bytes; return p; };
  __bf16* xb  = (__bf16*)carve((size_t)NE_X * 2);  // 16 MB
  __bf16* Wqb = (__bf16*)carve((size_t)NE_W * 2);  //  8 MB
  __bf16* Wkb = (__bf16*)carve((size_t)NE_W * 2);
  __bf16* Wvb = (__bf16*)carve((size_t)NE_W * 2);
  __bf16* Wob = (__bf16*)carve((size_t)NE_W * 2);
  float*  T32 = (float*)carve((size_t)NE_X * 4);   // 32 MB reused Q/K/V fp32
  __bf16* Qb  = (__bf16*)carve((size_t)NE_X * 2);
  __bf16* Kb  = (__bf16*)carve((size_t)NE_X * 2);
  __bf16* Vt  = (__bf16*)carve((size_t)NE_X * 2);
  __bf16* Ob  = (__bf16*)carve((size_t)NE_X * 2);
  (void)ws_size; (void)in_sizes; (void)n_in; (void)out_size;

  // 1. convert inputs to bf16
  f32_to_bf16<<<NE_X / 256, 256, 0, stream>>>(x,  xb,  NE_X);
  f32_to_bf16<<<NE_W / 256, 256, 0, stream>>>(Wq, Wqb, NE_W);
  f32_to_bf16<<<NE_W / 256, 256, 0, stream>>>(Wk, Wkb, NE_W);
  f32_to_bf16<<<NE_W / 256, 256, 0, stream>>>(Wv, Wvb, NE_W);
  f32_to_bf16<<<NE_W / 256, 256, 0, stream>>>(Wo, Wob, NE_W);

  dim3 ggrid(DM / 64, M / 128);   // (32, 32)
  dim3 gblk(128);

  // 2. Q projection + RoPE
  gemm_bf16_nt<<<ggrid, gblk, 0, stream>>>(xb, Wqb, T32, M, DM, DM);
  rope_to_bf16<<<(NE_X / 2) / 256, 256, 0, stream>>>(T32, Qb);
  // 3. K projection + RoPE (reuse T32)
  gemm_bf16_nt<<<ggrid, gblk, 0, stream>>>(xb, Wkb, T32, M, DM, DM);
  rope_to_bf16<<<(NE_X / 2) / 256, 256, 0, stream>>>(T32, Kb);
  // 4. V projection + transpose (reuse T32)
  gemm_bf16_nt<<<ggrid, gblk, 0, stream>>>(xb, Wvb, T32, M, DM, DM);
  v_transpose_bf16<<<NE_X / 256, 256, 0, stream>>>(T32, Vt);

  // 5. flash attention
  dim3 agrid(SEQ / 16, BATCH * NH);   // (128, 32)
  flash_attn<<<agrid, dim3(32), 0, stream>>>(Qb, Kb, Vt, Ob);

  // 6. output projection (fp32 result straight to d_out)
  gemm_bf16_nt<<<ggrid, gblk, 0, stream>>>(Ob, Wob, out, M, DM, DM);
}
